// Decoder_42906723287116
// MI455X (gfx1250) — matrix operand order repaired and checked by reading, not compiled, not run on hardware
//
#include <hip/hip_runtime.h>
#include <hip/hip_bf16.h>

#define BB 32
#define T_INN 384
#define T_STEPS 399
#define T_OUTT 400
#define N_MELL 80
#define ENCC 512
#define RNN 1024
#define PRENETT 256
#define PRENET_KP 96      // prenet input K padded 80 -> 96
#define ATT_DIMM 128
#define LOC_FF 32
#define LOC_KK 31
#define NEG_VAL -1000000000.0f

typedef __attribute__((ext_vector_type(16))) __bf16 v16bf;
typedef __attribute__((ext_vector_type(8)))  float v8f;
typedef __attribute__((ext_vector_type(4)))  unsigned int u32x4;

__device__ __forceinline__ unsigned short f2bf(float x) {
  unsigned u = __float_as_uint(x);
  u += 0x7FFFu + ((u >> 16) & 1u);
  return (unsigned short)(u >> 16);
}
__device__ __forceinline__ float sigm(float x) { return 1.f / (1.f + __expf(-x)); }

// ---------------- conversion / packing kernels ----------------

__global__ void k_cvt_bf16(const float* __restrict__ s, unsigned short* __restrict__ d, long n) {
  long i = (long)blockIdx.x * blockDim.x + threadIdx.x;
  if (i < n) d[i] = f2bf(s[i]);
}

// convert + pad columns: src N x Ks (f32) -> dst N x Kd (bf16, zero-padded)
__global__ void k_cvt_pad(const float* __restrict__ s, unsigned short* __restrict__ d,
                          int N, int Ks, int Kd) {
  int i = blockIdx.x * blockDim.x + threadIdx.x;
  if (i >= N * Kd) return;
  int n = i / Kd, k = i % Kd;
  d[i] = (k < Ks) ? f2bf(s[n * Ks + k]) : (unsigned short)0;
}

// column-concat two row-major f32 matrices (N x K0, N x K1) into bf16 N x (K0+K1)
__global__ void k_concat2_bf16(const float* __restrict__ s0, int K0,
                               const float* __restrict__ s1, int K1,
                               unsigned short* __restrict__ d, long n) {
  long i = (long)blockIdx.x * blockDim.x + threadIdx.x;
  if (i >= n) return;
  int K = K0 + K1;
  long r = i / K; int k = (int)(i % K);
  float v = (k < K0) ? s0[r * K0 + k] : s1[r * K1 + (k - K0)];
  d[i] = f2bf(v);
}

// pack proj_w (80x1536) + gate_w (1x1536) into 96x1536 bf16 (rows 81..95 zero)
__global__ void k_pack_pg(const float* __restrict__ proj, const float* __restrict__ gate,
                          unsigned short* __restrict__ d) {
  int i = blockIdx.x * blockDim.x + threadIdx.x;
  if (i >= 96 * 1536) return;
  int n = i / 1536, k = i % 1536;
  float v = (n < N_MELL) ? proj[n * 1536 + k] : (n == N_MELL ? gate[k] : 0.f);
  d[i] = f2bf(v);
}

__global__ void k_bias_sum(const float* __restrict__ a, const float* __restrict__ b,
                           float* __restrict__ d, int n) {
  int i = blockIdx.x * blockDim.x + threadIdx.x;
  if (i < n) d[i] = a[i] + b[i];
}

__global__ void k_bias_pg(const float* __restrict__ pb, const float* __restrict__ gb,
                          float* __restrict__ d) {
  int i = blockIdx.x * blockDim.x + threadIdx.x;
  if (i >= 96) return;
  d[i] = (i < N_MELL) ? pb[i] : (i == N_MELL ? gb[0] : 0.f);
}

// build prenet input (padded to 96 cols): din[t*32+b, m] = (t==0||m>=80) ? 0 : dec[b, m, t-1]
__global__ void k_din(const float* __restrict__ dec, unsigned short* __restrict__ d) {
  int i = blockIdx.x * blockDim.x + threadIdx.x;
  if (i >= T_STEPS * BB * PRENET_KP) return;
  int t = i / (BB * PRENET_KP);
  int r = i % (BB * PRENET_KP);
  int b = r / PRENET_KP, m = r % PRENET_KP;
  float v = (t == 0 || m >= N_MELL) ? 0.f : dec[((long)b * N_MELL + m) * T_OUTT + (t - 1)];
  d[i] = f2bf(v);
}

// ---------------- WMMA GEMM: C[MxN] = act( concat(A0,A1,A2)[MxK] @ W[NxK]^T + bias ) ----------------
// Requirements (all satisfied by callers): M, N multiples of 16; every K segment a multiple
// of 32; all row strides multiples of 8 elements; base pointers 16B-aligned.
// block = 128 (4 waves); each wave owns one 16x16 N-tile; blockIdx.x = M-tile.
union Frag { v16bf v; u32x4 q[2]; };

// accumulate one K-segment: straight-line loop, unrolled x2 so the scheduler can
// overlap the next tile's 4x global_load_b128 with the current tile's wait+wmma.
__device__ __forceinline__ v8f seg_accum(v8f acc,
                                         const unsigned short* __restrict__ As,
                                         long m, int lda, int h,
                                         const unsigned short* __restrict__ Wseg,
                                         int Kseg) {
  const unsigned short* __restrict__ pa_base = As + m * (long)lda + 16 * h;
  const unsigned short* __restrict__ pb_base = Wseg + 16 * h;
#pragma unroll 2
  for (int kk = 0; kk < Kseg; kk += 32) {
    Frag a, b;
    // A operand, contiguous k order used here (not the ISA table order): lane r holds row m; k = 16h..16h+15 (u[0..15])
    // -> two contiguous 16B loads (same k order as B).
    const u32x4* __restrict__ pa = (const u32x4*)(pa_base + kk);
    a.q[0] = pa[0];
    a.q[1] = pa[1];                  // +8 elements
    // B operand, same contiguous k order as A: lane r holds column n; k = kk+16h..kk+16h+15 -> 32B.
    const u32x4* __restrict__ pb = (const u32x4*)(pb_base + kk);
    b.q[0] = pb[0];
    b.q[1] = pb[1];
    acc = __builtin_amdgcn_wmma_f32_16x16x32_bf16(false, a.v, false, b.v, (short)0, acc, false, false);
  }
  return acc;
}

__global__ void k_wmma_gemm(const unsigned short* __restrict__ A0, int K0, int lda0,
                            const unsigned short* __restrict__ A1, int K1, int lda1,
                            const unsigned short* __restrict__ A2, int K2, int lda2,
                            const unsigned short* __restrict__ W,
                            const float* __restrict__ bias,
                            float* __restrict__ Cf, unsigned short* __restrict__ Cb,
                            int ldc, int N, int relu) {
  const int lane = threadIdx.x & 31;
  const int wave = threadIdx.x >> 5;
  const int tm = blockIdx.x;
  const int tn = blockIdx.y * 4 + wave;
  if (tn * 16 >= N) return;          // wave-uniform; EXEC stays all-1s for WMMA

  const int K = K0 + K1 + K2;
  const int r = lane & 15;
  const int h = lane >> 4;
  const long m = tm * 16 + r;        // A row for this lane
  const long n = tn * 16 + r;        // B column for this lane (row of W)
  const unsigned short* __restrict__ Wrow = W + n * (long)K;

  v8f acc = {0.f, 0.f, 0.f, 0.f, 0.f, 0.f, 0.f, 0.f};

  // per-segment straight-line k-loops (no per-iteration segment select)
  if (K0) acc = seg_accum(acc, A0, m, lda0, h, Wrow, K0);
  if (K1) acc = seg_accum(acc, A1, m, lda1, h, Wrow + K0, K1);
  if (K2) acc = seg_accum(acc, A2, m, lda2, h, Wrow + K0 + K1, K2);

  // C/D layout: VGPR i -> M = i + 8*h ; N = lane & 15
  const long nn = tn * 16 + (lane & 15);
  const float bv = bias ? bias[nn] : 0.f;
#pragma unroll
  for (int i = 0; i < 8; ++i) {
    long mm = tm * 16 + h * 8 + i;
    float v = acc[i] + bv;
    if (relu) v = fmaxf(v, 0.f);
    if (Cf) Cf[mm * ldc + nn] = v;
    if (Cb) Cb[mm * ldc + nn] = f2bf(v);
  }
}

// ---------------- LSTM pointwise update ----------------
__global__ void k_lstm_update(const float* __restrict__ gates,  // 32x4096 [i|f|g|o]
                              float* __restrict__ c, float* __restrict__ hf,
                              unsigned short* __restrict__ hb) {
  int idx = blockIdx.x * blockDim.x + threadIdx.x;
  if (idx >= BB * RNN) return;
  int b = idx >> 10, j = idx & (RNN - 1);
  const float* g = gates + (long)b * 4 * RNN;
  float i_ = sigm(g[j]);
  float f_ = sigm(g[RNN + j]);
  float gg = tanhf(g[2 * RNN + j]);
  float o_ = sigm(g[3 * RNN + j]);
  float c2 = f_ * c[idx] + i_ * gg;
  float hh = o_ * tanhf(c2);
  c[idx] = c2;
  hf[idx] = hh;
  hb[idx] = f2bf(hh);
}

// ---------------- fused location-sensitive attention step ----------------
// one block per batch element; blockDim = 384 (one thread per encoder time step)
__global__ void k_attention(const float* __restrict__ pq,      // 32x128
                            const float* __restrict__ pm,      // 32x384x128
                            const float* __restrict__ memory,  // 32x384x512
                            const int*   __restrict__ memlen,  // 32
                            const float* __restrict__ conv_w,  // 32x2x31
                            const float* __restrict__ ld_w,    // 128x32
                            const float* __restrict__ v_w,     // 128
                            float* __restrict__ aw, float* __restrict__ awc,  // 32x384
                            float* __restrict__ ctx, unsigned short* __restrict__ ctx_bf,
                            float* __restrict__ align_out, int t_step) {
  __shared__ float s_aw[T_INN], s_awc[T_INN], s_w[T_INN];
  __shared__ float s_red[512];
  __shared__ float s_ld[ATT_DIMM * LOC_FF];
  __shared__ float s_cw[LOC_FF * 2 * LOC_KK];
  __shared__ float s_v[ATT_DIMM];

  const int b = blockIdx.x;
  const int t = threadIdx.x;

  for (int i = t; i < ATT_DIMM * LOC_FF; i += T_INN) s_ld[i] = ld_w[i];
  for (int i = t; i < LOC_FF * 2 * LOC_KK; i += T_INN) s_cw[i] = conv_w[i];
  if (t < ATT_DIMM) s_v[t] = v_w[t];
  s_aw[t] = aw[b * T_INN + t];
  s_awc[t] = awc[b * T_INN + t];
  __syncthreads();

  // location conv: 2-ch x 31-tap, pad 15
  float locf[LOC_FF];
#pragma unroll
  for (int f = 0; f < LOC_FF; ++f) locf[f] = 0.f;
  for (int k = 0; k < LOC_KK; ++k) {
    int tt = t + k - 15;
    if (tt >= 0 && tt < T_INN) {
      float x0 = s_aw[tt], x1 = s_awc[tt];
#pragma unroll
      for (int f = 0; f < LOC_FF; ++f)
        locf[f] += x0 * s_cw[f * 62 + k] + x1 * s_cw[f * 62 + 31 + k];
    }
  }

  // energy = v . tanh(pq + loc_dense(loc) + processed_memory)
  float e = 0.f;
  const float* pqb = pq + b * ATT_DIMM;
  const float* pmb = pm + ((long)b * T_INN + t) * ATT_DIMM;
  for (int a = 0; a < ATT_DIMM; ++a) {
    float s = pqb[a] + pmb[a];
    const float* lw = &s_ld[a * LOC_FF];
#pragma unroll
    for (int f = 0; f < LOC_FF; ++f) s += locf[f] * lw[f];
    e += s_v[a] * tanhf(s);
  }
  if (t >= memlen[b]) e = NEG_VAL;

  // softmax over T_INN (padded reduce to 512)
  s_red[t] = e;
  if (t < 512 - T_INN) s_red[T_INN + t] = -3.0e38f;
  __syncthreads();
  for (int s = 256; s >= 1; s >>= 1) {
    if (t < s) s_red[t] = fmaxf(s_red[t], s_red[t + s]);
    __syncthreads();
  }
  float mx = s_red[0];
  __syncthreads();
  float ew = __expf(e - mx);
  s_red[t] = ew;
  if (t < 512 - T_INN) s_red[T_INN + t] = 0.f;
  __syncthreads();
  for (int s = 256; s >= 1; s >>= 1) {
    if (t < s) s_red[t] += s_red[t + s];
    __syncthreads();
  }
  float w = ew / s_red[0];

  s_w[t] = w;
  aw[b * T_INN + t] = w;
  awc[b * T_INN + t] = s_awc[t] + w;
  align_out[((long)b * T_STEPS + t_step) * T_INN + t] = w;
  __syncthreads();

  // context = w @ memory
  for (int e_ = t; e_ < ENCC; e_ += T_INN) {
    float acc = 0.f;
    const float* mb = memory + (long)b * T_INN * ENCC + e_;
    for (int tt = 0; tt < T_INN; ++tt) acc += s_w[tt] * mb[(long)tt * ENCC];
    ctx[b * ENCC + e_] = acc;
    ctx_bf[b * ENCC + e_] = f2bf(acc);
  }
}

// scatter projection results into strided outputs
__global__ void k_scatter(const float* __restrict__ projC,  // 32x96
                          float* __restrict__ out_mel, float* __restrict__ out_gate, int t) {
  int i = blockIdx.x * blockDim.x + threadIdx.x;
  if (i >= BB * 81) return;
  int b = i / 81, j = i % 81;
  float v = projC[b * 96 + j];
  if (j < N_MELL) out_mel[((long)b * N_MELL + j) * T_STEPS + t] = v;
  else out_gate[(long)b * T_STEPS + t] = v;
}

// ---------------- host orchestration ----------------

extern "C" void kernel_launch(void* const* d_in, const int* in_sizes, int n_in,
                              void* d_out, int out_size, void* d_ws, size_t ws_size,
                              hipStream_t stream) {
  const float* memory      = (const float*)d_in[0];
  const float* dec_inputs  = (const float*)d_in[1];
  const int*   memlen      = (const int*)d_in[2];
  const float* prenet_w1   = (const float*)d_in[3];
  const float* prenet_w2   = (const float*)d_in[4];
  const float* att_wih     = (const float*)d_in[5];
  const float* att_whh     = (const float*)d_in[6];
  const float* att_bih     = (const float*)d_in[7];
  const float* att_bhh     = (const float*)d_in[8];
  const float* query_w     = (const float*)d_in[9];
  const float* memory_w    = (const float*)d_in[10];
  const float* v_w         = (const float*)d_in[11];
  const float* loc_conv_w  = (const float*)d_in[12];
  const float* loc_dense_w = (const float*)d_in[13];
  const float* dec_wih     = (const float*)d_in[14];
  const float* dec_whh     = (const float*)d_in[15];
  const float* dec_bih     = (const float*)d_in[16];
  const float* dec_bhh     = (const float*)d_in[17];
  const float* proj_w      = (const float*)d_in[18];
  const float* proj_b      = (const float*)d_in[19];
  const float* gate_w      = (const float*)d_in[20];
  const float* gate_b      = (const float*)d_in[21];
  (void)in_sizes; (void)n_in; (void)out_size; (void)ws_size;

  float* out       = (float*)d_out;
  float* out_mel   = out;
  float* out_gate  = out + (long)BB * N_MELL * T_STEPS;
  float* out_align = out_gate + (long)BB * T_STEPS;

  size_t off = 0;
  auto alloc = [&](size_t bytes) -> void* {
    void* p = (char*)d_ws + off;
    off += (bytes + 255) & ~(size_t)255;
    return p;
  };

  unsigned short* W_att  = (unsigned short*)alloc((size_t)4096 * 1792 * 2);
  unsigned short* W_dec  = (unsigned short*)alloc((size_t)4096 * 2560 * 2);
  unsigned short* W_q    = (unsigned short*)alloc((size_t)128 * 1024 * 2);
  unsigned short* W_mem  = (unsigned short*)alloc((size_t)128 * 512 * 2);
  unsigned short* W_p1   = (unsigned short*)alloc((size_t)256 * PRENET_KP * 2);
  unsigned short* W_p2   = (unsigned short*)alloc((size_t)256 * 256 * 2);
  unsigned short* W_pg   = (unsigned short*)alloc((size_t)96 * 1536 * 2);
  unsigned short* din_bf = (unsigned short*)alloc((size_t)T_STEPS * BB * PRENET_KP * 2);
  unsigned short* pre1   = (unsigned short*)alloc((size_t)T_STEPS * BB * PRENETT * 2);
  unsigned short* xs_bf  = (unsigned short*)alloc((size_t)T_STEPS * BB * PRENETT * 2);
  unsigned short* mem_bf = (unsigned short*)alloc((size_t)BB * T_INN * ENCC * 2);
  float* pm       = (float*)alloc((size_t)BB * T_INN * ATT_DIMM * 4);
  float* gates    = (float*)alloc((size_t)BB * 4 * RNN * 4);
  float* pq       = (float*)alloc((size_t)BB * ATT_DIMM * 4);
  float* projC    = (float*)alloc((size_t)BB * 96 * 4);
  float* bias_att = (float*)alloc(4096 * 4);
  float* bias_dec = (float*)alloc(4096 * 4);
  float* bias_pg  = (float*)alloc(96 * 4);
  float* ah = (float*)alloc((size_t)BB * RNN * 4);
  float* ac = (float*)alloc((size_t)BB * RNN * 4);
  float* dh = (float*)alloc((size_t)BB * RNN * 4);
  float* dc = (float*)alloc((size_t)BB * RNN * 4);
  float* ctx = (float*)alloc((size_t)BB * ENCC * 4);
  float* aw  = (float*)alloc((size_t)BB * T_INN * 4);
  float* awc = (float*)alloc((size_t)BB * T_INN * 4);
  unsigned short* ah_bf  = (unsigned short*)alloc((size_t)BB * RNN * 2);
  unsigned short* dh_bf  = (unsigned short*)alloc((size_t)BB * RNN * 2);
  unsigned short* ctx_bf = (unsigned short*)alloc((size_t)BB * ENCC * 2);

  // ---- one-time conversions (recomputed every call; deterministic) ----
  {
    long n;
    n = (long)4096 * 1792;
    k_concat2_bf16<<<(unsigned)((n + 255) / 256), 256, 0, stream>>>(att_wih, 768, att_whh, 1024, W_att, n);
    n = (long)4096 * 2560;
    k_concat2_bf16<<<(unsigned)((n + 255) / 256), 256, 0, stream>>>(dec_wih, 1536, dec_whh, 1024, W_dec, n);
    n = 128 * 1024;  k_cvt_bf16<<<(unsigned)((n + 255) / 256), 256, 0, stream>>>(query_w, W_q, n);
    n = 128 * 512;   k_cvt_bf16<<<(unsigned)((n + 255) / 256), 256, 0, stream>>>(memory_w, W_mem, n);
    k_cvt_pad<<<(256 * PRENET_KP + 255) / 256, 256, 0, stream>>>(prenet_w1, W_p1, 256, N_MELL, PRENET_KP);
    n = 256 * 256;   k_cvt_bf16<<<(unsigned)((n + 255) / 256), 256, 0, stream>>>(prenet_w2, W_p2, n);
    k_pack_pg<<<(96 * 1536 + 255) / 256, 256, 0, stream>>>(proj_w, gate_w, W_pg);
    n = (long)BB * T_INN * ENCC;
    k_cvt_bf16<<<(unsigned)((n + 255) / 256), 256, 0, stream>>>(memory, mem_bf, n);
    k_din<<<(T_STEPS * BB * PRENET_KP + 255) / 256, 256, 0, stream>>>(dec_inputs, din_bf);
    k_bias_sum<<<16, 256, 0, stream>>>(att_bih, att_bhh, bias_att, 4096);
    k_bias_sum<<<16, 256, 0, stream>>>(dec_bih, dec_bhh, bias_dec, 4096);
    k_bias_pg<<<1, 96, 0, stream>>>(proj_b, gate_b, bias_pg);
  }

  // ---- zero initial state ----
  hipMemsetAsync(ah, 0, (size_t)BB * RNN * 4, stream);
  hipMemsetAsync(ac, 0, (size_t)BB * RNN * 4, stream);
  hipMemsetAsync(dh, 0, (size_t)BB * RNN * 4, stream);
  hipMemsetAsync(dc, 0, (size_t)BB * RNN * 4, stream);
  hipMemsetAsync(ctx, 0, (size_t)BB * ENCC * 4, stream);
  hipMemsetAsync(aw, 0, (size_t)BB * T_INN * 4, stream);
  hipMemsetAsync(awc, 0, (size_t)BB * T_INN * 4, stream);
  hipMemsetAsync(ah_bf, 0, (size_t)BB * RNN * 2, stream);
  hipMemsetAsync(dh_bf, 0, (size_t)BB * RNN * 2, stream);
  hipMemsetAsync(ctx_bf, 0, (size_t)BB * ENCC * 2, stream);

  auto gemm = [&](const unsigned short* A0, int K0, int lda0,
                  const unsigned short* A1, int K1, int lda1,
                  const unsigned short* A2, int K2, int lda2,
                  const unsigned short* W, const float* bias,
                  float* Cf, unsigned short* Cb, int ldc, int M, int N, int relu) {
    dim3 grid((unsigned)(M / 16), (unsigned)(((N + 15) / 16 + 3) / 4));
    k_wmma_gemm<<<grid, 128, 0, stream>>>(A0, K0, lda0, A1, K1, lda1, A2, K2, lda2,
                                          W, bias, Cf, Cb, ldc, N, relu);
  };

  // ---- precompute processed_memory = memory @ memory_w^T : (12288 x 128), K=512 ----
  gemm(mem_bf, ENCC, ENCC, nullptr, 0, 0, nullptr, 0, 0,
       W_mem, nullptr, pm, nullptr, ATT_DIMM, BB * T_INN, ATT_DIMM, 0);

  // ---- precompute prenet over all 399 steps: (12768 x 256) twice, relu ----
  gemm(din_bf, PRENET_KP, PRENET_KP, nullptr, 0, 0, nullptr, 0, 0,
       W_p1, nullptr, nullptr, pre1, PRENETT, T_STEPS * BB, PRENETT, 1);
  gemm(pre1, PRENETT, PRENETT, nullptr, 0, 0, nullptr, 0, 0,
       W_p2, nullptr, nullptr, xs_bf, PRENETT, T_STEPS * BB, PRENETT, 1);

  // ---- sequential decoder scan ----
  for (int t = 0; t < T_STEPS; ++t) {
    const unsigned short* xs_t = xs_bf + (size_t)t * BB * PRENETT;

    // attention LSTM gates: [x(256)|ctx(512)|ah(1024)] @ [wih|whh]^T -> 32x4096
    gemm(xs_t, PRENETT, PRENETT, ctx_bf, ENCC, ENCC, ah_bf, RNN, RNN,
         W_att, bias_att, gates, nullptr, 4 * RNN, BB, 4 * RNN, 0);
    k_lstm_update<<<(BB * RNN + 255) / 256, 256, 0, stream>>>(gates, ac, ah, ah_bf);

    // query projection: ah @ query_w^T -> 32x128
    gemm(ah_bf, RNN, RNN, nullptr, 0, 0, nullptr, 0, 0,
         W_q, nullptr, pq, nullptr, ATT_DIMM, BB, ATT_DIMM, 0);

    // fused location attention + softmax + context
    k_attention<<<BB, T_INN, 0, stream>>>(pq, pm, memory, memlen,
                                          loc_conv_w, loc_dense_w, v_w,
                                          aw, awc, ctx, ctx_bf, out_align, t);

    // decoder LSTM gates: [ah(1024)|ctx(512)|dh(1024)] @ [wih|whh]^T -> 32x4096
    gemm(ah_bf, RNN, RNN, ctx_bf, ENCC, ENCC, dh_bf, RNN, RNN,
         W_dec, bias_dec, gates, nullptr, 4 * RNN, BB, 4 * RNN, 0);
    k_lstm_update<<<(BB * RNN + 255) / 256, 256, 0, stream>>>(gates, dc, dh, dh_bf);

    // projection + gate: [dh|ctx] @ [proj;gate]^T -> 32x96
    gemm(dh_bf, RNN, RNN, ctx_bf, ENCC, ENCC, nullptr, 0, 0,
         W_pg, bias_pg, projC, nullptr, 96, BB, 96, 0);
    k_scatter<<<(BB * 81 + 127) / 128, 128, 0, stream>>>(projC, out_mel, out_gate, t);
  }
}
